// MoEBlock_49314814492978
// MI455X (gfx1250) — matrix operand order repaired and checked by reading, not compiled, not run on hardware
//
#include <hip/hip_runtime.h>

// MoE block, MI455X (gfx1250, wave32, WMMA bf16).
// v4: add global_prefetch_b8 (via __builtin_prefetch) on the strided
// prepacked-B streams, 4 k-chunks ahead, in both GEMM inner loops.

#define T_TOK 2048
#define IDIM  1024
#define HID   4096
#define NEXP  8

#define MAX_SLOTS 4224   // 2*T + padding headroom (16-aligned per expert)
#define MAX_TILES 264    // MAX_SLOTS/16

#define KC1 (IDIM / 32)  // 32  k-chunks in GEMM1
#define NT1 (HID / 16)   // 256 n-tiles  in GEMM1
#define KC2 (HID / 32)   // 128 k-chunks in GEMM2
#define NT2 (IDIM / 16)  // 64  n-tiles  in GEMM2

// workspace layout (indices in 4-byte units unless noted _BYTE)
#define OFF_CNT    0      // int[8]
#define OFF_PO     16     // int[9]
#define OFF_EID    64     // int[2T]
#define OFF_WT     8192   // float[2T]
#define OFF_TOK    16384  // int[MAX_SLOTS]
#define OFF_WSLOT  24576  // float[MAX_SLOTS]
#define OFF_SLOTOF 32768  // int[2T]
#define HBUF_BYTE  262144                                   // bf16[MAX_SLOTS*HID]  (34.6MB)
#define YBUF_BYTE  (HBUF_BYTE + MAX_SLOTS * HID * 2)        // f32 [MAX_SLOTS*IDIM] (17.3MB)
#define WP1_BYTE   (YBUF_BYTE + MAX_SLOTS * IDIM * 4)       // u32 prepacked W1 (64MB)
#define WP2_BYTE   (WP1_BYTE + NEXP * IDIM * HID * 2)       // u32 prepacked W2 (64MB)
#define XBF_BYTE   (WP2_BYTE + NEXP * HID * IDIM * 2)       // bf16[T*IDIM] (4MB)

typedef __attribute__((ext_vector_type(16))) __bf16 v16bf;
typedef __attribute__((ext_vector_type(8)))  float  v8f;

union V16 { v16bf v; unsigned u[8]; uint4 q[2]; };

static __device__ __forceinline__ unsigned bfr(float f) {
  unsigned x = __float_as_uint(f);
  return (x + 0x7fffu + ((x >> 16) & 1u)) >> 16;   // round-to-nearest-even bf16
}
static __device__ __forceinline__ unsigned pack2(float lo, float hi) {
  return bfr(lo) | (bfr(hi) << 16);
}

// ---------------------------------------------------------------- router
__global__ void k_router(const float* __restrict__ xs, const float* __restrict__ Wg,
                         int* __restrict__ eid, float* __restrict__ wt) {
  int t = blockIdx.x * blockDim.x + threadIdx.x;
  if (t >= T_TOK) return;
  float acc[NEXP];
#pragma unroll
  for (int e = 0; e < NEXP; ++e) acc[e] = 0.f;
  const float* x = xs + (size_t)t * IDIM;
  for (int d = 0; d < IDIM; d += 4) {
    float4 xv = *(const float4*)(x + d);
#pragma unroll
    for (int e = 0; e < NEXP; ++e) {
      float4 wv = *(const float4*)(Wg + (size_t)e * IDIM + d);
      acc[e] += xv.x * wv.x + xv.y * wv.y + xv.z * wv.z + xv.w * wv.w;
    }
  }
  int i0 = 0;
#pragma unroll
  for (int e = 1; e < NEXP; ++e) if (acc[e] > acc[i0]) i0 = e;
  int i1 = (i0 == 0) ? 1 : 0;
#pragma unroll
  for (int e = 0; e < NEXP; ++e) if (e != i0 && acc[e] > acc[i1]) i1 = e;
  float l0 = acc[i0], l1 = acc[i1];
  float m  = fmaxf(l0, l1);
  float p0 = __expf(l0 - m), p1 = __expf(l1 - m);
  float inv = 1.f / (p0 + p1);
  eid[2 * t]     = i0;  wt[2 * t]     = p0 * inv;
  eid[2 * t + 1] = i1;  wt[2 * t + 1] = p1 * inv;
}

// ------------------------------------------------- deterministic compaction
__global__ void k_assign(int* __restrict__ wsI, float* __restrict__ wsF) {
  if (threadIdx.x != 0 || blockIdx.x != 0) return;
  const int* eid = wsI + OFF_EID;
  int cnt[NEXP];
#pragma unroll
  for (int e = 0; e < NEXP; ++e) cnt[e] = 0;
  for (int i = 0; i < 2 * T_TOK; ++i) cnt[eid[i]]++;
  int po[NEXP + 1];
  po[0] = 0;
  for (int e = 0; e < NEXP; ++e) po[e + 1] = po[e] + ((cnt[e] + 15) & ~15);
  for (int e = 0; e < NEXP; ++e) { wsI[OFF_CNT + e] = cnt[e]; wsI[OFF_PO + e] = po[e]; }
  wsI[OFF_PO + NEXP] = po[NEXP];
  int cur[NEXP];
#pragma unroll
  for (int e = 0; e < NEXP; ++e) cur[e] = 0;
  for (int i = 0; i < 2 * T_TOK; ++i) {
    int e = eid[i];
    int slot = po[e] + cur[e]++;
    wsI[OFF_TOK + slot]   = i >> 1;
    wsF[OFF_WSLOT + slot] = wsF[OFF_WT + i];
    wsI[OFF_SLOTOF + i]   = slot;
  }
}

// ------------------------------------- weight prepack: fp32 -> bf16 B-layout
// dst[(((e*KC + kc)*NT + nt)*32 + lane)*8 + j] packs W[k][n],W[k+1][n]
// with lane = kh*16 + nl, n = nt*16 + nl, k = kc*32 + kh*16 + 2j.
__global__ void k_prep(const float* __restrict__ src, unsigned* __restrict__ dst,
                       int K, int N) {
  int id = blockIdx.x * blockDim.x + threadIdx.x;
  int NT = N >> 4, KC = K >> 5;
  int nl = id & 15;
  int j  = (id >> 4) & 7;
  int kh = (id >> 7) & 1;
  int tl = id >> 8;
  int nt = tl % NT;
  int rest = tl / NT;
  int kc = rest % KC;
  int e  = rest / KC;
  if (e >= NEXP) return;
  int n = nt * 16 + nl;
  int k = kc * 32 + kh * 16 + 2 * j;
  const float* s = src + (size_t)e * K * N + (size_t)k * N + n;
  unsigned p = pack2(s[0], s[N]);
  int lane = kh * 16 + nl;
  dst[((((size_t)e * KC + kc) * NT + nt) * 32 + lane) * 8 + j] = p;
}

// ------------------------------------------- activation prepack: fp32->bf16
__global__ void k_prepx(const float* __restrict__ src, unsigned* __restrict__ dst) {
  int id = blockIdx.x * blockDim.x + threadIdx.x;
  if (id >= T_TOK * IDIM / 2) return;
  float2 p = *(const float2*)(src + 2 * (size_t)id);
  dst[id] = pack2(p.x, p.y);
}

// ------------------------------------------------------------- GEMM1 (x@W1)
// grid (MAX_TILES, NT1/16), 128 threads (4 waves, 4 n-tiles each).
__launch_bounds__(128)
__global__ void k_gemm1(const unsigned short* __restrict__ xbf, const unsigned* __restrict__ Wp1,
                        const float* __restrict__ b1, const int* __restrict__ wsI,
                        const float* __restrict__ wsF, unsigned short* __restrict__ hbuf) {
  const int* po = wsI + OFF_PO;
  int base = blockIdx.x * 16;
  if (base >= po[NEXP]) return;
  int e = 0;
#pragma unroll
  for (int q = 0; q < NEXP; ++q) { if (base >= po[q + 1]) e = q + 1; }
  int vend = po[e] + wsI[OFF_CNT + e];

  __shared__ float wsh[16];
  if (threadIdx.x < 16) {
    int slot = base + threadIdx.x;
    wsh[threadIdx.x] = (slot < vend) ? wsF[OFF_WSLOT + slot] : 0.f;
  }
  __syncthreads();

  int lane = threadIdx.x & 31, wave = threadIdx.x >> 5;
  int mrow = lane & 15;
  int slot = base + mrow;
  int tok  = (slot < vend) ? wsI[OFF_TOK + slot] : 0;
  const unsigned short* arow = xbf + (size_t)tok * IDIM;
  int koff = (lane < 16) ? 0 : 16;
  int nt0  = blockIdx.y * 16 + wave * 4;

  const unsigned* Bb[4];
#pragma unroll
  for (int nt = 0; nt < 4; ++nt)
    Bb[nt] = Wp1 + ((((size_t)e * KC1) * NT1 + (nt0 + nt)) * 32 + lane) * 8;

  v8f c[4];
#pragma unroll
  for (int nt = 0; nt < 4; ++nt)
#pragma unroll
    for (int r = 0; r < 8; ++r) c[nt][r] = 0.f;

  V16 A0, A1, B0[4], B1[4];
  {
    const unsigned short* ak = arow + koff;
    A0.q[0] = *(const uint4*)(ak);
    A0.q[1] = *(const uint4*)(ak + 8);
#pragma unroll
    for (int nt = 0; nt < 4; ++nt) {
      const unsigned* bp = Bb[nt];
      B0[nt].q[0] = *(const uint4*)(bp);
      B0[nt].q[1] = *(const uint4*)(bp + 4);
    }
  }
  for (int kc = 0; kc < KC1; kc += 2) {
    int kp = (kc + 4 < KC1) ? kc + 4 : KC1 - 1;   // prefetch distance 4
#pragma unroll
    for (int nt = 0; nt < 4; ++nt)
      __builtin_prefetch(Bb[nt] + (size_t)kp * (NT1 * 256), 0, 3);
    {  // prefetch kc+1 into regs
      const unsigned short* ak = arow + (kc + 1) * 32 + koff;
      A1.q[0] = *(const uint4*)(ak);
      A1.q[1] = *(const uint4*)(ak + 8);
#pragma unroll
      for (int nt = 0; nt < 4; ++nt) {
        const unsigned* bp = Bb[nt] + (size_t)(kc + 1) * (NT1 * 256);
        B1[nt].q[0] = *(const uint4*)(bp);
        B1[nt].q[1] = *(const uint4*)(bp + 4);
      }
    }
#pragma unroll
    for (int nt = 0; nt < 4; ++nt)
      c[nt] = __builtin_amdgcn_wmma_f32_16x16x32_bf16(false, A0.v, false, B0[nt].v,
                                                      (short)0, c[nt], false, false);
    int k2 = (kc + 2 < KC1) ? kc + 2 : kc;   // clamp; junk never consumed
    {  // prefetch kc+2 into regs
      const unsigned short* ak = arow + k2 * 32 + koff;
      A0.q[0] = *(const uint4*)(ak);
      A0.q[1] = *(const uint4*)(ak + 8);
#pragma unroll
      for (int nt = 0; nt < 4; ++nt) {
        const unsigned* bp = Bb[nt] + (size_t)k2 * (NT1 * 256);
        B0[nt].q[0] = *(const uint4*)(bp);
        B0[nt].q[1] = *(const uint4*)(bp + 4);
      }
    }
#pragma unroll
    for (int nt = 0; nt < 4; ++nt)
      c[nt] = __builtin_amdgcn_wmma_f32_16x16x32_bf16(false, A1.v, false, B1[nt].v,
                                                      (short)0, c[nt], false, false);
  }
  int mhi = (lane < 16) ? 0 : 8;
#pragma unroll
  for (int nt = 0; nt < 4; ++nt) {
    int n = (nt0 + nt) * 16 + mrow;
    float bias = b1[(size_t)e * HID + n];
#pragma unroll
    for (int r = 0; r < 8; ++r) {
      int m = mhi + r;
      float v = fmaxf(c[nt][r] + bias, 0.f) * wsh[m];   // relu + gate weight
      hbuf[(size_t)(base + m) * HID + n] = (unsigned short)bfr(v);
    }
  }
}

// ------------------------------------------------------------- GEMM2 (h@W2)
// grid (MAX_TILES, NT2/16), 128 threads.
__launch_bounds__(128)
__global__ void k_gemm2(const unsigned short* __restrict__ hbuf, const unsigned* __restrict__ Wp2,
                        const float* __restrict__ b2, const int* __restrict__ wsI,
                        const float* __restrict__ wsF, float* __restrict__ ybuf) {
  const int* po = wsI + OFF_PO;
  int base = blockIdx.x * 16;
  if (base >= po[NEXP]) return;
  int e = 0;
#pragma unroll
  for (int q = 0; q < NEXP; ++q) { if (base >= po[q + 1]) e = q + 1; }
  int vend = po[e] + wsI[OFF_CNT + e];

  __shared__ float wsh[16];
  if (threadIdx.x < 16) {
    int slot = base + threadIdx.x;
    wsh[threadIdx.x] = (slot < vend) ? wsF[OFF_WSLOT + slot] : 0.f;
  }
  __syncthreads();

  int lane = threadIdx.x & 31, wave = threadIdx.x >> 5;
  int mrow = lane & 15;
  const unsigned short* arow = hbuf + (size_t)(base + mrow) * HID;
  int koff = (lane < 16) ? 0 : 16;
  int nt0  = blockIdx.y * 16 + wave * 4;

  const unsigned* Bb[4];
#pragma unroll
  for (int nt = 0; nt < 4; ++nt)
    Bb[nt] = Wp2 + ((((size_t)e * KC2) * NT2 + (nt0 + nt)) * 32 + lane) * 8;

  v8f c[4];
#pragma unroll
  for (int nt = 0; nt < 4; ++nt)
#pragma unroll
    for (int r = 0; r < 8; ++r) c[nt][r] = 0.f;

  V16 A0, A1, B0[4], B1[4];
  {
    const unsigned short* ak = arow + koff;
    A0.q[0] = *(const uint4*)(ak);
    A0.q[1] = *(const uint4*)(ak + 8);
#pragma unroll
    for (int nt = 0; nt < 4; ++nt) {
      const unsigned* bp = Bb[nt];
      B0[nt].q[0] = *(const uint4*)(bp);
      B0[nt].q[1] = *(const uint4*)(bp + 4);
    }
  }
  for (int kc = 0; kc < KC2; kc += 2) {
    int kp = (kc + 4 < KC2) ? kc + 4 : KC2 - 1;   // prefetch distance 4
#pragma unroll
    for (int nt = 0; nt < 4; ++nt)
      __builtin_prefetch(Bb[nt] + (size_t)kp * (NT2 * 256), 0, 3);
    {  // prefetch kc+1 into regs
      const unsigned short* ak = arow + (kc + 1) * 32 + koff;
      A1.q[0] = *(const uint4*)(ak);
      A1.q[1] = *(const uint4*)(ak + 8);
#pragma unroll
      for (int nt = 0; nt < 4; ++nt) {
        const unsigned* bp = Bb[nt] + (size_t)(kc + 1) * (NT2 * 256);
        B1[nt].q[0] = *(const uint4*)(bp);
        B1[nt].q[1] = *(const uint4*)(bp + 4);
      }
    }
#pragma unroll
    for (int nt = 0; nt < 4; ++nt)
      c[nt] = __builtin_amdgcn_wmma_f32_16x16x32_bf16(false, A0.v, false, B0[nt].v,
                                                      (short)0, c[nt], false, false);
    int k2 = (kc + 2 < KC2) ? kc + 2 : kc;   // clamp; junk never consumed
    {  // prefetch kc+2 into regs
      const unsigned short* ak = arow + k2 * 32 + koff;
      A0.q[0] = *(const uint4*)(ak);
      A0.q[1] = *(const uint4*)(ak + 8);
#pragma unroll
      for (int nt = 0; nt < 4; ++nt) {
        const unsigned* bp = Bb[nt] + (size_t)k2 * (NT2 * 256);
        B0[nt].q[0] = *(const uint4*)(bp);
        B0[nt].q[1] = *(const uint4*)(bp + 4);
      }
    }
#pragma unroll
    for (int nt = 0; nt < 4; ++nt)
      c[nt] = __builtin_amdgcn_wmma_f32_16x16x32_bf16(false, A1.v, false, B1[nt].v,
                                                      (short)0, c[nt], false, false);
  }
  int mhi = (lane < 16) ? 0 : 8;
#pragma unroll
  for (int nt = 0; nt < 4; ++nt) {
    int n = (nt0 + nt) * 16 + mrow;
    float bias = b2[(size_t)e * IDIM + n];
#pragma unroll
    for (int r = 0; r < 8; ++r) {
      int m = mhi + r;
      ybuf[(size_t)(base + m) * IDIM + n] = c[nt][r] + wsh[m] * bias;
    }
  }
}

// ------------------------------------------------------------------ combine
__global__ void k_combine(const float* __restrict__ ybuf, const int* __restrict__ slot_of,
                          float* __restrict__ out) {
  int idx = blockIdx.x * blockDim.x + threadIdx.x;
  if (idx >= T_TOK * IDIM) return;
  int t = idx >> 10, d = idx & (IDIM - 1);
  int s0 = slot_of[2 * t], s1 = slot_of[2 * t + 1];
  out[idx] = ybuf[(size_t)s0 * IDIM + d] + ybuf[(size_t)s1 * IDIM + d];
}

extern "C" void kernel_launch(void* const* d_in, const int* in_sizes, int n_in,
                              void* d_out, int out_size, void* d_ws, size_t ws_size,
                              hipStream_t stream) {
  const float* xs = (const float*)d_in[0];
  // d_in[1] = top_k scalar (hardwired to 2)
  const float* Wg = (const float*)d_in[2];
  const float* W1 = (const float*)d_in[3];
  const float* b1 = (const float*)d_in[4];
  const float* W2 = (const float*)d_in[5];
  const float* b2 = (const float*)d_in[6];
  float* out = (float*)d_out;

  int*   wsI  = (int*)d_ws;
  float* wsF  = (float*)d_ws;
  unsigned short* hbuf = (unsigned short*)((char*)d_ws + HBUF_BYTE);
  float*          ybuf = (float*)((char*)d_ws + YBUF_BYTE);
  unsigned*       Wp1  = (unsigned*)((char*)d_ws + WP1_BYTE);
  unsigned*       Wp2  = (unsigned*)((char*)d_ws + WP2_BYTE);
  unsigned*       xbf  = (unsigned*)((char*)d_ws + XBF_BYTE);

  int n1 = NEXP * (IDIM * HID / 2);   // u32 count
  int n2 = NEXP * (HID * IDIM / 2);
  k_prep<<<n1 / 256, 256, 0, stream>>>(W1, Wp1, IDIM, HID);
  k_prep<<<n2 / 256, 256, 0, stream>>>(W2, Wp2, HID, IDIM);
  k_prepx<<<(T_TOK * IDIM / 2) / 256, 256, 0, stream>>>(xs, xbf);

  k_router<<<T_TOK / 256, 256, 0, stream>>>(xs, Wg, wsI + OFF_EID, wsF + OFF_WT);
  k_assign<<<1, 32, 0, stream>>>(wsI, wsF);
  k_gemm1<<<dim3(MAX_TILES, NT1 / 16), 128, 0, stream>>>((const unsigned short*)xbf, Wp1, b1,
                                                         wsI, wsF, hbuf);
  k_gemm2<<<dim3(MAX_TILES, NT2 / 16), 128, 0, stream>>>(hbuf, Wp2, b2, wsI, wsF, ybuf);
  k_combine<<<(T_TOK * IDIM) / 256, 256, 0, stream>>>(ybuf, wsI + OFF_SLOTOF, out);
}
